// LM_63926293233954
// MI455X (gfx1250) — matrix operand order repaired and checked by reading, not compiled, not run on hardware
//
#include <hip/hip_runtime.h>
#include <hip/hip_bf16.h>

typedef __attribute__((ext_vector_type(16))) __bf16 v16bf;
typedef __attribute__((ext_vector_type(8)))  __bf16 v8bf;
typedef __attribute__((ext_vector_type(8)))  float  v8f;

#define BM 128
#define BN 128
#define BK 32

static __device__ __forceinline__ unsigned pk2(float a, float b)
{
    unsigned short lo = __builtin_bit_cast(unsigned short, (__bf16)a);
    unsigned short hi = __builtin_bit_cast(unsigned short, (__bf16)b);
    return (unsigned)lo | ((unsigned)hi << 16);
}

// ---------------------------------------------------------------------------
// Generic batched GEMM: C = act(A @ B + bias) + resid
// A: [M,K] row-major (lda); B: [K,N] (ldb) or, if BT, layout B[n*ldb+k];
// C: [M,N] (ldc). fp32 in/out, bf16 WMMA accumulate-in-f32.
// Double-buffered LDS pipeline: global loads of tile k+1 overlap the 8 WMMAs
// of tile k; one workgroup barrier per K-step. K must be a multiple of 32.
// ALIGNED instantiation drops all M/N bounds checks (tile-exact launches).
// ---------------------------------------------------------------------------
template <bool BT, bool AL>
__global__ __launch_bounds__(256) void gemm_bf16_wmma(
    const float* __restrict__ A, long long lda, long long strideA,
    const float* __restrict__ B, long long ldb, long long strideB,
    float* __restrict__ C, long long ldc, long long strideC,
    const float* __restrict__ bias,
    const float* __restrict__ resid,
    int M, int N, int K, int do_relu)
{
    __shared__ __bf16 As[2][BM][BK + 8];   // [m][k], K-contiguous
    __shared__ __bf16 Bs[2][BN][BK + 8];   // [n][k], K-contiguous

    const int bz = blockIdx.z;
    A += (long long)bz * strideA;
    B += (long long)bz * strideB;
    C += (long long)bz * strideC;
    if (resid) resid += (long long)bz * strideC;

    const int tile_n0 = blockIdx.x * BN;
    const int tile_m0 = blockIdx.y * BM;

    const int tid  = threadIdx.x;
    const int lane = tid & 31;
    const int wave = tid >> 5;   // 0..7
    const int wm   = wave & 1;   // 2 wave-rows (64 rows each)
    const int wn   = wave >> 1;  // 4 wave-cols (32 cols each)
    const int half = lane >> 4;  // 0/1
    const int l16  = lane & 15;

    v8f acc[4][2] = {};
    float4 ra[4], rb[4];

    // ---- stage K-tile kt from global into registers ----
    auto load_tile = [&](int kt) {
        const int k0 = kt * BK;
        #pragma unroll
        for (int it = 0; it < 4; ++it) {
            int idx = tid + it * 256;
            int r   = idx >> 3;
            int c4  = (idx & 7) << 2;
            int gm  = tile_m0 + r;
            ra[it] = make_float4(0.f, 0.f, 0.f, 0.f);
            if (AL || gm < M)
                ra[it] = *reinterpret_cast<const float4*>(&A[(long long)gm * lda + k0 + c4]);
        }
        if (!BT) {
            #pragma unroll
            for (int it = 0; it < 4; ++it) {
                int idx = tid + it * 256;
                int r   = idx >> 5;          // k row 0..31
                int c4  = (idx & 31) << 2;   // n offset
                int gn  = tile_n0 + c4;
                rb[it] = make_float4(0.f, 0.f, 0.f, 0.f);
                if (AL || gn < N)
                    rb[it] = *reinterpret_cast<const float4*>(&B[(long long)(k0 + r) * ldb + gn]);
            }
        } else {
            #pragma unroll
            for (int it = 0; it < 4; ++it) {
                int idx = tid + it * 256;
                int nl  = idx >> 3;          // n 0..127
                int c4  = (idx & 7) << 2;    // k offset
                int gn  = tile_n0 + nl;
                rb[it] = make_float4(0.f, 0.f, 0.f, 0.f);
                if (AL || gn < N)
                    rb[it] = *reinterpret_cast<const float4*>(&B[(long long)gn * ldb + k0 + c4]);
            }
        }
    };

    // ---- convert registers -> bf16 and store into LDS buffer `buf` ----
    auto store_tile = [&](int buf) {
        #pragma unroll
        for (int it = 0; it < 4; ++it) {
            int idx = tid + it * 256;
            int r   = idx >> 3;
            int c4  = (idx & 7) << 2;
            *reinterpret_cast<uint2*>(&As[buf][r][c4]) =
                make_uint2(pk2(ra[it].x, ra[it].y), pk2(ra[it].z, ra[it].w));
        }
        if (!BT) {
            #pragma unroll
            for (int it = 0; it < 4; ++it) {
                int idx = tid + it * 256;
                int r   = idx >> 5;
                int c4  = (idx & 31) << 2;
                Bs[buf][c4 + 0][r] = (__bf16)rb[it].x;
                Bs[buf][c4 + 1][r] = (__bf16)rb[it].y;
                Bs[buf][c4 + 2][r] = (__bf16)rb[it].z;
                Bs[buf][c4 + 3][r] = (__bf16)rb[it].w;
            }
        } else {
            #pragma unroll
            for (int it = 0; it < 4; ++it) {
                int idx = tid + it * 256;
                int nl  = idx >> 3;
                int c4  = (idx & 7) << 2;
                *reinterpret_cast<uint2*>(&Bs[buf][nl][c4]) =
                    make_uint2(pk2(rb[it].x, rb[it].y), pk2(rb[it].z, rb[it].w));
            }
        }
    };

    // ---- 8 WMMAs on LDS buffer `buf` ----
    auto compute = [&](int buf) {
        v16bf vb[2];
        #pragma unroll
        for (int sn = 0; sn < 2; ++sn) {
            int col = wn * 32 + sn * 16 + l16;
            v8bf b0 = *reinterpret_cast<const v8bf*>(&Bs[buf][col][half * 8]);
            v8bf b1 = *reinterpret_cast<const v8bf*>(&Bs[buf][col][16 + half * 8]);
            vb[sn] = __builtin_shufflevector(b0, b1,
                0, 1, 2, 3, 4, 5, 6, 7, 8, 9, 10, 11, 12, 13, 14, 15);
        }
        #pragma unroll
        for (int sm = 0; sm < 4; ++sm) {
            int row = wm * 64 + sm * 16 + l16;
            v8bf a0 = *reinterpret_cast<const v8bf*>(&As[buf][row][half * 8]);
            v8bf a1 = *reinterpret_cast<const v8bf*>(&As[buf][row][16 + half * 8]);
            v16bf va = __builtin_shufflevector(a0, a1,
                0, 1, 2, 3, 4, 5, 6, 7, 8, 9, 10, 11, 12, 13, 14, 15);
            #pragma unroll
            for (int sn = 0; sn < 2; ++sn) {
                acc[sm][sn] = __builtin_amdgcn_wmma_f32_16x16x32_bf16(
                    false, va, false, vb[sn], (short)0, acc[sm][sn], false, false);
            }
        }
    };

    const int nk = K / BK;
    load_tile(0);
    store_tile(0);
    __syncthreads();
    for (int kt = 0; kt < nk; ++kt) {
        const int cur = kt & 1;
        if (kt + 1 < nk) load_tile(kt + 1);
        compute(cur);
        if (kt + 1 < nk) {
            store_tile(cur ^ 1);
            __syncthreads();
        }
    }

    // ---- epilogue ----
    #pragma unroll
    for (int sm = 0; sm < 4; ++sm) {
        #pragma unroll
        for (int sn = 0; sn < 2; ++sn) {
            int n = tile_n0 + wn * 32 + sn * 16 + l16;
            if (!AL && n >= N) continue;
            float bv = bias ? bias[n] : 0.f;
            #pragma unroll
            for (int r = 0; r < 8; ++r) {
                int m = tile_m0 + wm * 64 + sm * 16 + r + 8 * half;
                if (!AL && m >= M) continue;
                float v = acc[sm][sn][r] + bv;
                if (do_relu) v = v > 0.f ? v : 0.f;
                if (resid) v += resid[(long long)m * ldc + n];
                C[(long long)m * ldc + n] = v;
            }
        }
    }
}

// ---------------------------------------------------------------------------
// Elementwise / reduction kernels
// ---------------------------------------------------------------------------
__global__ void embed_kernel(const int* __restrict__ ids,
                             const float* __restrict__ E,
                             float* __restrict__ x)
{
    long long row = blockIdx.x;
    long long src = (long long)ids[row] * 1024;
    float4 f = *reinterpret_cast<const float4*>(&E[src + threadIdx.x * 4]);
    *reinterpret_cast<float4*>(&x[row * 1024 + threadIdx.x * 4]) = f;
}

__global__ __launch_bounds__(256) void ln_kernel(const float* __restrict__ x,
                                                 float* __restrict__ out,
                                                 const float* __restrict__ g,
                                                 const float* __restrict__ b)
{
    const int D = 1024;
    long long row = blockIdx.x;
    int tid = threadIdx.x;
    float4 v = *reinterpret_cast<const float4*>(&x[row * D + tid * 4]);
    __shared__ float r1[256], r2[256];
    r1[tid] = v.x + v.y + v.z + v.w;
    r2[tid] = v.x * v.x + v.y * v.y + v.z * v.z + v.w * v.w;
    __syncthreads();
    for (int off = 128; off > 0; off >>= 1) {
        if (tid < off) { r1[tid] += r1[tid + off]; r2[tid] += r2[tid + off]; }
        __syncthreads();
    }
    float mean = r1[0] / D;
    float var  = r2[0] / D - mean * mean;
    float inv  = rsqrtf(var + 1e-5f);
    float4 o;
    o.x = (v.x - mean) * inv * g[tid * 4 + 0] + b[tid * 4 + 0];
    o.y = (v.y - mean) * inv * g[tid * 4 + 1] + b[tid * 4 + 1];
    o.z = (v.z - mean) * inv * g[tid * 4 + 2] + b[tid * 4 + 2];
    o.w = (v.w - mean) * inv * g[tid * 4 + 3] + b[tid * 4 + 3];
    *reinterpret_cast<float4*>(&out[row * D + tid * 4]) = o;
}

__global__ __launch_bounds__(256) void knorm_kernel(float* __restrict__ qkv)
{
    long long row = blockIdx.x;
    int tid = threadIdx.x;
    float* k = qkv + row * 3072 + 1024;
    float4 v = *reinterpret_cast<const float4*>(&k[tid * 4]);
    __shared__ float r2[256];
    r2[tid] = v.x * v.x + v.y * v.y + v.z * v.z + v.w * v.w;
    __syncthreads();
    for (int off = 128; off > 0; off >>= 1) {
        if (tid < off) r2[tid] += r2[tid + off];
        __syncthreads();
    }
    float inv = rsqrtf(r2[0]);
    v.x *= inv; v.y *= inv; v.z *= inv; v.w *= inv;
    *reinterpret_cast<float4*>(&k[tid * 4]) = v;
}

__global__ __launch_bounds__(256) void lr_kernel(const float* __restrict__ h,
                                                 const float* __restrict__ wlr,
                                                 const float* __restrict__ blr,
                                                 float* __restrict__ lr)
{
    long long row = blockIdx.x;
    int tid = threadIdx.x;
    float4 hv = *reinterpret_cast<const float4*>(&h[row * 1024 + tid * 4]);
    float4 wv = *reinterpret_cast<const float4*>(&wlr[tid * 4]);
    __shared__ float r1[256];
    r1[tid] = hv.x * wv.x + hv.y * wv.y + hv.z * wv.z + hv.w * wv.w;
    __syncthreads();
    for (int off = 128; off > 0; off >>= 1) {
        if (tid < off) r1[tid] += r1[tid + off];
        __syncthreads();
    }
    if (tid == 0) lr[row] = r1[0] + blr[0];
}

__global__ void wmul_kernel(const float* __restrict__ u,
                            const float* __restrict__ lr,
                            float* __restrict__ w, long long n)
{
    long long i = (long long)blockIdx.x * blockDim.x + threadIdx.x;
    if (i < n) w[i] = u[i] * lr[i >> 10];
}

// ---------------------------------------------------------------------------
// In-block unit-lower-triangular solve: 32 rows, one column per thread.
// The 32x32 KK diagonal block is staged into LDS via the gfx1250 async
// global->LDS path (one b32 per thread), tracked with ASYNCcnt.
// ---------------------------------------------------------------------------
__global__ __launch_bounds__(1024) void solve_block_kernel(
    const float* __restrict__ KK,   // [nb][1024][1024]
    const float* __restrict__ qkv,  // v at col offset 2048, row stride 3072
    const float* __restrict__ tmp,  // [nb][32][1024] off-diagonal partial
    float* __restrict__ u,          // [nb][1024][1024]
    int blk)
{
    const int T = 1024, D = 1024;
    int nb = blockIdx.x;
    int d  = threadIdx.x;
    int t0 = blk * 32;
    KK += (long long)nb * T * T;

    __shared__ float kk[32][32];
    {
        int i = threadIdx.x;                       // 1024 threads, 1024 elems
        unsigned long long gaddr =
            (unsigned long long)(const void*)&KK[(long long)(t0 + (i >> 5)) * T + t0 + (i & 31)];
        unsigned int laddr = (unsigned int)(unsigned long long)(void*)&kk[i >> 5][i & 31];
        asm volatile("global_load_async_to_lds_b32 %0, %1, off"
                     :: "v"(laddr), "v"(gaddr) : "memory");
        asm volatile("s_wait_asynccnt 0" ::: "memory");
    }
    __syncthreads();

    float uloc[32];
    #pragma unroll
    for (int r = 0; r < 32; ++r) {
        long long grow = (long long)nb * T + t0 + r;
        float val = qkv[grow * 3072 + 2048 + d];
        if (blk > 0) val -= tmp[((long long)nb * 32 + r) * D + d];
        #pragma unroll
        for (int j = 0; j < r; ++j) val -= kk[r][j] * uloc[j];
        uloc[r] = val;
        u[grow * D + d] = val;
    }
}

// ---------------------------------------------------------------------------
// Host orchestration
// ---------------------------------------------------------------------------
static void launch_gemm(hipStream_t s,
                        const float* A, long long lda, long long sA,
                        const float* B, long long ldb, long long sB,
                        float* C, long long ldc, long long sC,
                        const float* bias, const float* resid,
                        int M, int N, int K, int batch, int btrans, int relu)
{
    dim3 grid((N + BN - 1) / BN, (M + BM - 1) / BM, batch);
    bool al = (M % BM == 0) && (N % BN == 0);
    if (btrans) {
        if (al) gemm_bf16_wmma<true, true><<<grid, 256, 0, s>>>(
                    A, lda, sA, B, ldb, sB, C, ldc, sC, bias, resid, M, N, K, relu);
        else    gemm_bf16_wmma<true, false><<<grid, 256, 0, s>>>(
                    A, lda, sA, B, ldb, sB, C, ldc, sC, bias, resid, M, N, K, relu);
    } else {
        if (al) gemm_bf16_wmma<false, true><<<grid, 256, 0, s>>>(
                    A, lda, sA, B, ldb, sB, C, ldc, sC, bias, resid, M, N, K, relu);
        else    gemm_bf16_wmma<false, false><<<grid, 256, 0, s>>>(
                    A, lda, sA, B, ldb, sB, C, ldc, sC, bias, resid, M, N, K, relu);
    }
}

extern "C" void kernel_launch(void* const* d_in, const int* in_sizes, int n_in,
                              void* d_out, int out_size, void* d_ws, size_t ws_size,
                              hipStream_t stream)
{
    (void)in_sizes; (void)n_in; (void)out_size; (void)ws_size;
    const int NB = 4, T = 1024, D = 1024, L = 4, V = 32000;
    const long long R = (long long)NB * T;          // 4096 rows

    const int*   ids = (const int*)  d_in[0];
    const float* E   = (const float*)d_in[1];
    const float* Wg  = (const float*)d_in[2];
    const float* bg  = (const float*)d_in[3];
    const float* Wlr = (const float*)d_in[4];
    const float* blr = (const float*)d_in[5];
    const float* Wo  = (const float*)d_in[6];
    const float* bo  = (const float*)d_in[7];
    const float* g_d = (const float*)d_in[8];
    const float* b_d = (const float*)d_in[9];
    const float* W1  = (const float*)d_in[10];
    const float* b1  = (const float*)d_in[11];
    const float* W2  = (const float*)d_in[12];
    const float* b2  = (const float*)d_in[13];
    const float* g_m = (const float*)d_in[14];
    const float* b_m = (const float*)d_in[15];
    const float* g_o = (const float*)d_in[16];
    const float* b_o = (const float*)d_in[17];
    float* out = (float*)d_out;

    float* ws = (float*)d_ws;
    float* x     = ws; ws += R * D;
    float* h     = ws; ws += R * D;
    float* qkv   = ws; ws += R * 3 * D;              // also reused as MLP t1
    float* KK    = ws; ws += (long long)NB * T * T;
    float* u     = ws; ws += (long long)NB * T * D;
    float* wbuf  = ws; ws += R * D;
    float* ybuf  = ws; ws += R * D;
    float* score = ws; ws += (long long)NB * T * T;
    float* tmp   = ws; ws += (long long)NB * 32 * D;
    float* lrb   = ws; ws += R;

    // x = E[ids]
    embed_kernel<<<dim3((unsigned)R), 256, 0, stream>>>(ids, E, x);

    for (int l = 0; l < L; ++l) {
        const float* Wg_l  = Wg  + (long long)l * D * 3 * D;
        const float* bg_l  = bg  + (long long)l * 3 * D;
        const float* Wlr_l = Wlr + (long long)l * D;
        const float* Wo_l  = Wo  + (long long)l * D * D;
        const float* W1_l  = W1  + (long long)l * D * D;
        const float* W2_l  = W2  + (long long)l * D * D;

        // --- Delta block ---
        ln_kernel<<<dim3((unsigned)R), 256, 0, stream>>>(x, h, g_d + l * D, b_d + l * D);
        // qkv = h @ Wg + bg
        launch_gemm(stream, h, D, 0, Wg_l, 3 * D, 0, qkv, 3 * D, 0,
                    bg_l, nullptr, (int)R, 3 * D, D, 1, 0, 0);
        // k /= ||k||
        knorm_kernel<<<dim3((unsigned)R), 256, 0, stream>>>(qkv);
        // lr = h @ Wlr + blr
        lr_kernel<<<dim3((unsigned)R), 256, 0, stream>>>(h, Wlr_l, blr + l, lrb);
        // KK = k k^T  (batched, B transposed)
        launch_gemm(stream, qkv + 1024, 3 * D, (long long)T * 3 * D,
                    qkv + 1024, 3 * D, (long long)T * 3 * D,
                    KK, T, (long long)T * T,
                    nullptr, nullptr, T, T, D, NB, 1, 0);
        // blocked unit-lower triangular solve: (I + stril(KK)) u = v
        for (int blk = 0; blk < T / 32; ++blk) {
            if (blk > 0) {
                launch_gemm(stream, KK + (long long)blk * 32 * T, T, (long long)T * T,
                            u, D, (long long)T * D,
                            tmp, D, 32LL * D,
                            nullptr, nullptr, 32, D, blk * 32, NB, 0, 0);
            }
            solve_block_kernel<<<dim3(NB), 1024, 0, stream>>>(KK, qkv, tmp, u, blk);
        }
        // score = q k^T (full, batched, B transposed)
        launch_gemm(stream, qkv, 3 * D, (long long)T * 3 * D,
                    qkv + 1024, 3 * D, (long long)T * 3 * D,
                    score, T, (long long)T * T,
                    nullptr, nullptr, T, T, D, NB, 1, 0);
        // w = lr * u
        wmul_kernel<<<dim3((unsigned)((R * D + 255) / 256)), 256, 0, stream>>>(
            u, lrb, wbuf, R * D);
        // y = score @ w  (batched)
        launch_gemm(stream, score, T, (long long)T * T,
                    wbuf, D, (long long)T * D,
                    ybuf, D, (long long)T * D,
                    nullptr, nullptr, T, D, T, NB, 0, 0);
        // x = x + y @ Wo + bo
        launch_gemm(stream, ybuf, D, 0, Wo_l, D, 0, x, D, 0,
                    bo + l * D, x, (int)R, D, D, 1, 0, 0);

        // --- MLP block ---
        ln_kernel<<<dim3((unsigned)R), 256, 0, stream>>>(x, h, g_m + l * D, b_m + l * D);
        // t1 = relu(h @ W1 + b1)  (reuse qkv buffer)
        launch_gemm(stream, h, D, 0, W1_l, D, 0, qkv, D, 0,
                    b1 + l * D, nullptr, (int)R, D, D, 1, 0, 1);
        // x = x + t1 @ W2 + b2
        launch_gemm(stream, qkv, D, 0, W2_l, D, 0, x, D, 0,
                    b2 + l * D, x, (int)R, D, D, 1, 0, 0);
    }

    // final LN + tied decoder: out = ln(x) @ E^T
    ln_kernel<<<dim3((unsigned)R), 256, 0, stream>>>(x, h, g_o, b_o);
    launch_gemm(stream, h, D, 0, E, D, 0, out, V, 0,
                nullptr, nullptr, (int)R, V, D, 1, 1, 0);
}
